// TransformerEncoderLayer_11132555231586
// MI455X (gfx1250) — matrix operand order repaired and checked by reading, not compiled, not run on hardware
//
#include <hip/hip_runtime.h>
#include <hip/hip_bf16.h>
#include <stdint.h>

typedef __attribute__((ext_vector_type(8)))  __bf16 v8bf;
typedef __attribute__((ext_vector_type(16))) __bf16 v16bf;
typedef __attribute__((ext_vector_type(8)))  float  v8f;

#define LOG2E 1.44269504088896340736f

static __device__ __forceinline__ v16bf bf_combine(v8bf lo, v8bf hi) {
    return __builtin_shufflevector(lo, hi, 0,1,2,3,4,5,6,7,8,9,10,11,12,13,14,15);
}

static __device__ __forceinline__ v8f wmma_bf16(v16bf a, v16bf b, v8f c) {
    // D = A(16x32 bf16) * B(32x16 bf16) + C(16x16 f32)
    return __builtin_amdgcn_wmma_f32_16x16x32_bf16(false, a, false, b, (short)0, c, false, false);
}

// ---------------------------------------------------------------------------
// CDNA5 async global->LDS copy (ASYNCcnt-tracked) via inline asm.
// VDST operand = LDS byte address (low 32 bits of the generic pointer, per the
// aperture rules); VADDR = 64-bit global address.
// ---------------------------------------------------------------------------
static __device__ __forceinline__ void cp_async_b128(void* lds, const void* g) {
    unsigned int l = (unsigned int)(uintptr_t)lds;
    asm volatile("global_load_async_to_lds_b128 %0, %1, off"
                 :
                 : "v"(l), "v"(g)
                 : "memory");
}

static __device__ __forceinline__ void cp_async_wait() {
    asm volatile("s_wait_asynccnt 0x0" ::: "memory");
}

// ---------------------------------------------------------------------------
// fp32 -> bf16 elementwise cast (vectorized)
// ---------------------------------------------------------------------------
__global__ __launch_bounds__(256) void cast_bf16_kernel(const float* __restrict__ in,
                                                        __bf16* __restrict__ out, int n) {
    int idx = (blockIdx.x * 256 + threadIdx.x) * 4;
    if (idx + 3 < n) {
        float4 f = *(const float4*)&in[idx];
        out[idx + 0] = (__bf16)f.x;
        out[idx + 1] = (__bf16)f.y;
        out[idx + 2] = (__bf16)f.z;
        out[idx + 3] = (__bf16)f.w;
    }
}

// ---------------------------------------------------------------------------
// fp32 [K][N] -> bf16 [N][K] (transpose + cast) via LDS tile
// ---------------------------------------------------------------------------
__global__ __launch_bounds__(256) void trans_cast_kernel(const float* __restrict__ in,
                                                         __bf16* __restrict__ out,
                                                         int K, int N) {
    __shared__ __bf16 tile[32][33];
    int x  = blockIdx.x * 32 + threadIdx.x;   // N index
    int y0 = blockIdx.y * 32;                 // K base
#pragma unroll
    for (int i = 0; i < 32; i += 8)
        tile[threadIdx.y + i][threadIdx.x] = (__bf16)in[(size_t)(y0 + threadIdx.y + i) * N + x];
    __syncthreads();
    int k = y0 + threadIdx.x;
#pragma unroll
    for (int i = 0; i < 32; i += 8)
        out[(size_t)(blockIdx.x * 32 + threadIdx.y + i) * K + k] = tile[threadIdx.x][threadIdx.y + i];
}

// ---------------------------------------------------------------------------
// WMMA GEMM:  C[M x N] (f32 acc) = A_bf16[M x K] row-major * BT_bf16[N x K] + bias
// block = 256 thr (8 waves), tile 128(M) x 128(N), K staged 32-wide,
// DOUBLE-BUFFERED LDS with async global->LDS copies overlapping WMMA.
// wave (wr=w&3, wc=w>>2) computes a 32 x 64 strip: 2 A-frags x 4 B-frags,
// 8 WMMAs per k-step.
// epilogue modes:
//   0: f32 row-major        1: bf16 [B,H,S,Dk] head layout (Q/K)
//   2: bf16 [B,H,Dk,S]      3: bf16 relu row-major
// ---------------------------------------------------------------------------
__global__ __launch_bounds__(256) void gemm_bf16_kernel(const __bf16* __restrict__ A,
                                                        const __bf16* __restrict__ BT,
                                                        const float* __restrict__ bias,
                                                        void* __restrict__ outp,
                                                        int M, int N, int K, int mode) {
    __shared__ __align__(16) __bf16 As[2][128][32];
    __shared__ __align__(16) __bf16 Bs[2][128][32];

    const int tid   = threadIdx.x;
    const int lane  = tid & 31;
    const int w     = tid >> 5;
    const int wr    = w & 3;
    const int wc    = w >> 2;
    const int mLane = lane & 15;
    const int g     = lane >> 4;
    const int rowBase = blockIdx.y * 128;
    const int colBase = blockIdx.x * 128;

    const v8f vzero = {0.f,0.f,0.f,0.f,0.f,0.f,0.f,0.f};
    v8f acc[2][4] = {{vzero, vzero, vzero, vzero}, {vzero, vzero, vzero, vzero}};

    // this thread's two 16B staging chunks (512 chunks cover 128x32 bf16)
    const int ch0 = tid * 2, ch1 = tid * 2 + 1;
    const int r0 = ch0 >> 2, c0 = (ch0 & 3) * 8;
    const int r1 = ch1 >> 2, c1 = (ch1 & 3) * 8;

    auto issue = [&](int buf, int k0) {
        cp_async_b128(&As[buf][r0][c0], &A [(size_t)(rowBase + r0) * K + k0 + c0]);
        cp_async_b128(&As[buf][r1][c1], &A [(size_t)(rowBase + r1) * K + k0 + c1]);
        cp_async_b128(&Bs[buf][r0][c0], &BT[(size_t)(colBase + r0) * K + k0 + c0]);
        cp_async_b128(&Bs[buf][r1][c1], &BT[(size_t)(colBase + r1) * K + k0 + c1]);
    };

    issue(0, 0);
    int cur = 0;
    for (int k0 = 0; k0 < K; k0 += 32) {
        cp_async_wait();          // my async copies for buf[cur] have landed
        __syncthreads();          // everyone's have; prev compute on buf[cur^1] done
        if (k0 + 32 < K) issue(cur ^ 1, k0 + 32);   // prefetch next tile (overlaps WMMA)

        v16bf af[2];
#pragma unroll
        for (int r = 0; r < 2; ++r) {
            const __bf16* arow = &As[cur][32 * wr + 16 * r + mLane][0];
            af[r] = bf_combine(*(const v8bf*)(arow + 16 * g),
                               *(const v8bf*)(arow + 16 * g + 8));
        }
#pragma unroll
        for (int t = 0; t < 4; ++t) {
            const __bf16* brow = &Bs[cur][wc * 64 + t * 16 + mLane][0];
            v16bf bf = bf_combine(*(const v8bf*)(brow + 16 * g),
                                  *(const v8bf*)(brow + 16 * g + 8));
            acc[0][t] = wmma_bf16(af[0], bf, acc[0][t]);
            acc[1][t] = wmma_bf16(af[1], bf, acc[1][t]);
        }
        cur ^= 1;
    }

#pragma unroll
    for (int t = 0; t < 4; ++t) {
        const int gcol = colBase + wc * 64 + t * 16 + mLane;
        const float bv = bias[gcol];
#pragma unroll
        for (int r = 0; r < 2; ++r) {
#pragma unroll
            for (int j = 0; j < 8; ++j) {
                const int grow = rowBase + 32 * wr + 16 * r + j + 8 * g;
                float v = acc[r][t][j] + bv;
                if (mode == 0) {
                    ((float*)outp)[(size_t)grow * N + gcol] = v;
                } else if (mode == 3) {
                    ((__bf16*)outp)[(size_t)grow * N + gcol] = (__bf16)fmaxf(v, 0.f);
                } else {
                    const int b = grow >> 11, s = grow & 2047;   // S = 2048
                    const int h = gcol >> 6,  d = gcol & 63;     // Dk = 64
                    if (mode == 1)
                        ((__bf16*)outp)[((size_t)(b * 16 + h) * 2048 + s) * 64 + d] = (__bf16)v;
                    else
                        ((__bf16*)outp)[((size_t)(b * 16 + h) * 64 + d) * 2048 + s] = (__bf16)v;
                }
            }
        }
    }
}

// ---------------------------------------------------------------------------
// Flash attention: one wave per 16-query tile, 8 waves/block (128 queries).
// 32 keys/step; K-tile + V^T-tile double-buffered in LDS via async copies;
// online softmax with 16-lane butterflies; P transposed through LDS to the
// WMMA A layout.
// ---------------------------------------------------------------------------
__global__ __launch_bounds__(256) void attn_kernel(const __bf16* __restrict__ Qh,
                                                   const __bf16* __restrict__ Kh,
                                                   const __bf16* __restrict__ VhT,
                                                   __bf16* __restrict__ ctx) {
    constexpr int S = 2048, DK = 64;
    __shared__ __align__(16) __bf16 Ks[2][32][64];
    __shared__ __align__(16) __bf16 Vs[2][64][32];
    __shared__ __align__(16) __bf16 Ps[8][16][32];

    const int nq = S / 128;                 // 16 q-blocks per (b,h)
    const int bh = blockIdx.x / nq;         // 0..31 = b*16+h
    const int qb = blockIdx.x % nq;
    const int h  = bh & 15;
    const int b  = bh >> 4;
    const int tid = threadIdx.x, lane = tid & 31, w = tid >> 5;
    const int mLane = lane & 15, g = lane >> 4;
    const int q0 = qb * 128 + w * 16;

    const __bf16* Qp = Qh  + (size_t)bh * S * DK;
    const __bf16* Kp = Kh  + (size_t)bh * S * DK;
    const __bf16* Vp = VhT + (size_t)bh * DK * S;

    // Q fragments (16 x 64 = two 16x32 A fragments), held in registers
    v16bf qf[2];
    {
        const __bf16* qrow = Qp + (size_t)(q0 + mLane) * DK;
#pragma unroll
        for (int c = 0; c < 2; ++c)
            qf[c] = bf_combine(*(const v8bf*)(qrow + c * 32 + 16 * g),
                               *(const v8bf*)(qrow + c * 32 + 16 * g + 8));
    }

    const v8f vzero = {0.f,0.f,0.f,0.f,0.f,0.f,0.f,0.f};
    v8f ctxacc[4] = {vzero, vzero, vzero, vzero};
    float mprev[8], lsum[8];
#pragma unroll
    for (int j = 0; j < 8; ++j) { mprev[j] = -3.0e38f; lsum[j] = 0.f; }

    const float SCL = 0.125f * LOG2E;       // 1/sqrt(64) folded with log2(e)

    const int kr = tid >> 3, kc = (tid & 7) * 8;   // K-tile staging coords
    const int vr = tid >> 2, vc = (tid & 3) * 8;   // V-tile staging coords
    auto issue = [&](int buf, int k0) {
        cp_async_b128(&Ks[buf][kr][kc], &Kp[(size_t)(k0 + kr) * DK + kc]);
        cp_async_b128(&Vs[buf][vr][vc], &Vp[(size_t)vr * S + k0 + vc]);
    };

    issue(0, 0);
    int cur = 0;
    for (int k0 = 0; k0 < S; k0 += 32) {
        cp_async_wait();
        __syncthreads();
        if (k0 + 32 < S) issue(cur ^ 1, k0 + 32);

        // scores: two 16x16 tiles, kdim = 64 via 2 WMMAs each
        v8f s0 = vzero, s1 = vzero;
#pragma unroll
        for (int c = 0; c < 2; ++c) {
            const __bf16* kr0 = &Ks[cur][mLane][c * 32 + 16 * g];
            v16bf b0 = bf_combine(*(const v8bf*)kr0, *(const v8bf*)(kr0 + 8));
            s0 = wmma_bf16(qf[c], b0, s0);
            const __bf16* kr1 = &Ks[cur][16 + mLane][c * 32 + 16 * g];
            v16bf b1 = bf_combine(*(const v8bf*)kr1, *(const v8bf*)(kr1 + 8));
            s1 = wmma_bf16(qf[c], b1, s1);
        }

        // online softmax, per accumulator row j (row = j + 8g, col = mLane)
#pragma unroll
        for (int j = 0; j < 8; ++j) {
            float a0 = s0[j] * SCL, a1 = s1[j] * SCL;
            float tm = fmaxf(a0, a1);
#pragma unroll
            for (int msk = 1; msk < 16; msk <<= 1)
                tm = fmaxf(tm, __shfl_xor(tm, msk, 16));
            float mn = fmaxf(mprev[j], tm);
            float sc = exp2f(mprev[j] - mn);
            float p0 = exp2f(a0 - mn);
            float p1 = exp2f(a1 - mn);
            float rs = p0 + p1;
#pragma unroll
            for (int msk = 1; msk < 16; msk <<= 1)
                rs += __shfl_xor(rs, msk, 16);
            lsum[j] = lsum[j] * sc + rs;
            mprev[j] = mn;
#pragma unroll
            for (int t = 0; t < 4; ++t) ctxacc[t][j] *= sc;
            Ps[w][j + 8 * g][mLane]      = (__bf16)p0;
            Ps[w][j + 8 * g][16 + mLane] = (__bf16)p1;
        }
        __syncthreads();

        // ctx += P(16x32) @ V(32x64)
        const __bf16* prow = &Ps[w][mLane][0];
        v16bf pf = bf_combine(*(const v8bf*)(prow + 16 * g),
                              *(const v8bf*)(prow + 16 * g + 8));
#pragma unroll
        for (int t = 0; t < 4; ++t) {
            const __bf16* vrp = &Vs[cur][t * 16 + mLane][16 * g];
            v16bf vf = bf_combine(*(const v8bf*)vrp, *(const v8bf*)(vrp + 8));
            ctxacc[t] = wmma_bf16(pf, vf, ctxacc[t]);
        }
        cur ^= 1;
    }

    // finalize: divide by row sum, write ctx in [B*S, 1024] row-major bf16
#pragma unroll
    for (int t = 0; t < 4; ++t) {
        const int col = h * 64 + t * 16 + mLane;
#pragma unroll
        for (int j = 0; j < 8; ++j) {
            const int row = b * S + q0 + j + 8 * g;
            ctx[(size_t)row * 1024 + col] = (__bf16)(ctxacc[t][j] / lsum[j]);
        }
    }
}

// ---------------------------------------------------------------------------
// out = LayerNorm(A + R) * gamma + beta ; fp32 out, optional bf16 copy
// one block (256 thr) per 1024-wide row
// ---------------------------------------------------------------------------
__global__ __launch_bounds__(256) void add_ln_kernel(const float* __restrict__ A,
                                                     const float* __restrict__ R,
                                                     const float* __restrict__ gamma,
                                                     const float* __restrict__ beta,
                                                     float* __restrict__ outf,
                                                     __bf16* __restrict__ outb,
                                                     int writeBf) {
    __shared__ float red[256];
    const int tid = threadIdx.x;
    const size_t base = (size_t)blockIdx.x * 1024;
    float v[4];
    float s = 0.f;
#pragma unroll
    for (int i = 0; i < 4; ++i) {
        int c = tid + i * 256;
        v[i] = A[base + c] + R[base + c];
        s += v[i];
    }
    red[tid] = s; __syncthreads();
    for (int o = 128; o > 0; o >>= 1) { if (tid < o) red[tid] += red[tid + o]; __syncthreads(); }
    float mu = red[0] * (1.f / 1024.f);
    __syncthreads();
    float sq = 0.f;
#pragma unroll
    for (int i = 0; i < 4; ++i) { float d = v[i] - mu; sq += d * d; }
    red[tid] = sq; __syncthreads();
    for (int o = 128; o > 0; o >>= 1) { if (tid < o) red[tid] += red[tid + o]; __syncthreads(); }
    float rstd = rsqrtf(red[0] * (1.f / 1024.f) + 1e-5f);
#pragma unroll
    for (int i = 0; i < 4; ++i) {
        int c = tid + i * 256;
        float y = (v[i] - mu) * rstd * gamma[c] + beta[c];
        outf[base + c] = y;
        if (writeBf) outb[base + c] = (__bf16)y;
    }
}

// ---------------------------------------------------------------------------
extern "C" void kernel_launch(void* const* d_in, const int* in_sizes, int n_in,
                              void* d_out, int out_size, void* d_ws, size_t ws_size,
                              hipStream_t stream) {
    constexpr int B = 2, S = 2048, DM = 1024, DF = 4096;
    constexpr int M = B * S;                       // 4096 rows

    const float* x  = (const float*)d_in[0];
    const float* Wq = (const float*)d_in[1];  const float* bq = (const float*)d_in[2];
    const float* Wk = (const float*)d_in[3];  const float* bk = (const float*)d_in[4];
    const float* Wv = (const float*)d_in[5];  const float* bv = (const float*)d_in[6];
    const float* Wo = (const float*)d_in[7];  const float* bo = (const float*)d_in[8];
    const float* W1 = (const float*)d_in[9];  const float* b1 = (const float*)d_in[10];
    const float* W2 = (const float*)d_in[11]; const float* b2 = (const float*)d_in[12];
    const float* g1 = (const float*)d_in[13]; const float* be1 = (const float*)d_in[14];
    const float* g2 = (const float*)d_in[15]; const float* be2 = (const float*)d_in[16];

    char* ws = (char*)d_ws;
    // byte offsets (all multiples of 256); total footprint ~104 MiB
    const size_t SZ_ACT_BF = (size_t)M * DM * 2;          // 8 MiB
    const size_t SZ_W_BF   = (size_t)DM * DM * 2;         // 2 MiB
    const size_t SZ_WF_BF  = (size_t)DM * DF * 2;         // 8 MiB
    const size_t SZ_ACT_F  = (size_t)M * DM * 4;          // 16 MiB

    size_t off = 0;
    __bf16* xb   = (__bf16*)(ws + off); off += SZ_ACT_BF;
    __bf16* WqT  = (__bf16*)(ws + off); off += SZ_W_BF;
    __bf16* WkT  = (__bf16*)(ws + off); off += SZ_W_BF;
    __bf16* WvT  = (__bf16*)(ws + off); off += SZ_W_BF;
    __bf16* WoT  = (__bf16*)(ws + off); off += SZ_W_BF;
    __bf16* W1T  = (__bf16*)(ws + off); off += SZ_WF_BF;
    __bf16* W2T  = (__bf16*)(ws + off); off += SZ_WF_BF;
    __bf16* Qhp  = (__bf16*)(ws + off); off += SZ_ACT_BF; // Qh..ctx span 32 MiB,
    __bf16* Khp  = (__bf16*)(ws + off); off += SZ_ACT_BF; // reused later for ffb
    __bf16* VhTp = (__bf16*)(ws + off); off += SZ_ACT_BF;
    __bf16* ctxb = (__bf16*)(ws + off); off += SZ_ACT_BF;
    float*  attn = (float*)(ws + off);  off += SZ_ACT_F;  // reused later for ffn out
    float*  h1   = (float*)(ws + off);  off += SZ_ACT_F;
    __bf16* h1b  = (__bf16*)(ws + off); off += SZ_ACT_BF;
    __bf16* ffb  = Qhp;                                   // 32 MiB alias (Qh..ctx)
    float*  ffn  = attn;                                  // 16 MiB alias
    (void)ws_size; (void)in_sizes; (void)n_in; (void)out_size;

    // 1. casts / weight transposes
    cast_bf16_kernel<<<(M * DM) / (256 * 4), 256, 0, stream>>>(x, xb, M * DM);
    dim3 tb(32, 8);
    trans_cast_kernel<<<dim3(DM / 32, DM / 32), tb, 0, stream>>>(Wq, WqT, DM, DM);
    trans_cast_kernel<<<dim3(DM / 32, DM / 32), tb, 0, stream>>>(Wk, WkT, DM, DM);
    trans_cast_kernel<<<dim3(DM / 32, DM / 32), tb, 0, stream>>>(Wv, WvT, DM, DM);
    trans_cast_kernel<<<dim3(DM / 32, DM / 32), tb, 0, stream>>>(Wo, WoT, DM, DM);
    trans_cast_kernel<<<dim3(DF / 32, DM / 32), tb, 0, stream>>>(W1, W1T, DM, DF);
    trans_cast_kernel<<<dim3(DM / 32, DF / 32), tb, 0, stream>>>(W2, W2T, DF, DM);

    // 2. QKV projections (WMMA)
    gemm_bf16_kernel<<<dim3(DM / 128, M / 128), 256, 0, stream>>>(xb, WqT, bq, Qhp,  M, DM, DM, 1);
    gemm_bf16_kernel<<<dim3(DM / 128, M / 128), 256, 0, stream>>>(xb, WkT, bk, Khp,  M, DM, DM, 1);
    gemm_bf16_kernel<<<dim3(DM / 128, M / 128), 256, 0, stream>>>(xb, WvT, bv, VhTp, M, DM, DM, 2);

    // 3. flash attention (WMMA)
    attn_kernel<<<B * 16 * (S / 128), 256, 0, stream>>>(Qhp, Khp, VhTp, ctxb);

    // 4. output projection + residual LayerNorm
    gemm_bf16_kernel<<<dim3(DM / 128, M / 128), 256, 0, stream>>>(ctxb, WoT, bo, attn, M, DM, DM, 0);
    add_ln_kernel<<<M, 256, 0, stream>>>(x, attn, g1, be1, h1, h1b, 1);

    // 5. FFN + residual LayerNorm
    gemm_bf16_kernel<<<dim3(DF / 128, M / 128), 256, 0, stream>>>(h1b, W1T, b1, ffb, M, DF, DM, 3);
    gemm_bf16_kernel<<<dim3(DM / 128, M / 128), 256, 0, stream>>>(ffb, W2T, b2, ffn, M, DM, DF, 0);
    add_ln_kernel<<<M, 256, 0, stream>>>(h1, ffn, g2, be2, (float*)d_out, (__bf16*)nullptr, 0);
}
